// ConsistentPointSelect_7834020348448
// MI455X (gfx1250) — compile-verified, matrix operand order checked by reading, not run on hardware
//
#include <hip/hip_runtime.h>

typedef __attribute__((ext_vector_type(16))) _Float16 v16h;
typedef __attribute__((ext_vector_type(8)))  _Float16 v8h;
typedef __attribute__((ext_vector_type(8)))  float    v8f;
typedef int v4i __attribute__((vector_size(16)));
typedef unsigned long long u64;
typedef unsigned int       u32;

#define BB     8
#define NN     50000
#define CC     128
#define KK     25000
#define NPAD   65536
#define NTILES 3125        // 50000 / 16, exact

// ---- workspace layout (bytes) ----
#define WS_B1   0                       // _Float16[4*4*512]  fc1 B-fragments
#define WS_B2   16384                   // _Float16[4*4*512]  fc2 B-fragments
#define WS_B3   32768                   // _Float16[6*8*512]  conv B-fragments
#define WS_FC34 81920                   // float[128]: fc3_w,fc3_b,fc4_w,fc4_b
#define WS_BNF  82432                   // float[256]: bn scale[128], shift[128]
#define WS_B12  83456                   // float[128]: fc1_b[64], fc2_b[64]
#define WS_KEYS 84480                   // u64[8*65536] sort keys

// ---- output layout (float elements) ----
#define OUT_W    ((size_t)0)
#define OUT_IDX  ((size_t)(BB*NN))                   // 400000 (int32 bits)
#define OUT_KPF  (OUT_IDX + (size_t)BB*KK)           // 600000
#define OUT_KRP  (OUT_KPF + (size_t)BB*KK*CC)        // 26,200,000
#define OUT_KNF  (OUT_KRP + (size_t)BB*KK*3)         // 26,800,000
#define OUT_KRN  (OUT_KNF + (size_t)BB*KK*CC)        // 52,400,000

#if __has_builtin(__builtin_amdgcn_global_load_async_to_lds_b128) && \
    __has_builtin(__builtin_amdgcn_s_wait_asynccnt)
#define USE_ASYNC_LDS 1
typedef __attribute__((address_space(1))) v4i* gv4i_p;
typedef __attribute__((address_space(3))) v4i* lv4i_p;
#endif

// ===================== weight prep: swizzle to WMMA B-fragments =====================
// B-fragment layout (16-bit, 32x16 KxN): lane L holds column N=L%16,
// halves j: K = (L<16?0:8) + (j<8 ? j : j+8), per 32-K step.
__global__ void prep_kernel(const float* __restrict__ fc1_w, const float* __restrict__ fc1_b,
                            const float* __restrict__ fc2_w, const float* __restrict__ fc2_b,
                            const float* __restrict__ fc3_w, const float* __restrict__ fc3_b,
                            const float* __restrict__ fc4_w, const float* __restrict__ fc4_b,
                            const float* __restrict__ conv_w, const float* __restrict__ conv_b,
                            const float* __restrict__ bn_g, const float* __restrict__ bn_be,
                            const float* __restrict__ bn_mu, const float* __restrict__ bn_var,
                            char* __restrict__ ws)
{
    _Float16* wAll = (_Float16*)(ws + WS_B1);   // B1 | B2 | B3 contiguous
    float*    fc34 = (float*)(ws + WS_FC34);
    float*    bnf  = (float*)(ws + WS_BNF);
    float*    b12  = (float*)(ws + WS_B12);
    int tid = threadIdx.x;

    for (int i = tid; i < 80 * 512; i += 256) {
        int frag = i >> 9;
        int e    = i & 511;
        int lane = e >> 4;
        int h    = e & 15;
        int kin  = ((lane < 16) ? 0 : 8) + ((h < 8) ? h : (h + 8));  // K within 32-step
        int nn   = lane & 15;
        float val;
        if (frag < 16) {                    // fc1_w [128][64]
            int ks = frag >> 2, t = frag & 3;
            val = fc1_w[(ks * 32 + kin) * 64 + t * 16 + nn];
        } else if (frag < 32) {             // fc2_w [128][64]
            int f = frag - 16, ks = f >> 2, t = f & 3;
            val = fc2_w[(ks * 32 + kin) * 64 + t * 16 + nn];
        } else {                            // conv_w [128][192], B = conv_w^T
            int f = frag - 32, ks = f >> 3, t = f & 7;
            val = conv_w[(t * 16 + nn) * 192 + ks * 32 + kin];
        }
        wAll[i] = (_Float16)val;
    }
    if (tid < 128) {
        int j = tid & 31;
        float v;
        if (tid < 32)       v = fc3_w[j];
        else if (tid < 64)  v = fc3_b[j];
        else if (tid < 96)  v = fc4_w[j];
        else                v = fc4_b[j];
        fc34[tid] = v;
        float s = bn_g[tid] * rsqrtf(bn_var[tid] + 1e-5f);
        bnf[tid]       = s;
        bnf[128 + tid] = (conv_b[tid] - bn_mu[tid]) * s + bn_be[tid];
        b12[tid] = (tid < 64) ? fc1_b[tid] : fc2_b[tid - 64];
    }
}

__global__ void padinit_kernel(u64* __restrict__ keys)
{
    int i = blockIdx.x * blockDim.x + threadIdx.x;
    const int per = NPAD - NN;
    if (i >= BB * per) return;
    int b = i / per, n = NN + (i - b * per);
    keys[(size_t)b * NPAD + n] = 0ull;   // sorts last (all real keys > 0)
}

// ===================== fused scoring kernel (WMMA) =====================
__launch_bounds__(256)
__global__ void score_kernel(const float* __restrict__ pointfea,
                             const float* __restrict__ normalfea,
                             const float* __restrict__ point,
                             const float* __restrict__ normal,
                             const char*  __restrict__ ws,
                             float* __restrict__ out_weight,
                             u64*   __restrict__ keys)
{
    const float* fc34 = (const float*)(ws + WS_FC34);
    const float* bnf  = (const float*)(ws + WS_BNF);
    const float* b12  = (const float*)(ws + WS_B12);

    __shared__ __align__(32) _Float16 sFeat[8][16 * 192];   // 48 KB feature staging
    __shared__ __align__(16) _Float16 sW[40960];            // 80 KB: B1|B2|B3 fragments

    int lane = threadIdx.x & 31;
    int wave = threadIdx.x >> 5;
    int tile = blockIdx.x * 8 + wave;
    int b    = blockIdx.y;

    // ---- stage all weight fragments (80 KB) global -> LDS, once per block ----
    {
        const char* gsrc = ws + WS_B1;
#ifdef USE_ASYNC_LDS
        unsigned lbase = (unsigned)(unsigned long long)(const void*)&sW[0];
        unsigned long long gbase = (unsigned long long)(const void*)gsrc;
        for (int i = threadIdx.x; i < 81920 / 16; i += 256) {
            gv4i_p g = (gv4i_p)(gbase + (unsigned long long)i * 16u);
            lv4i_p l = (lv4i_p)(lbase + (unsigned)i * 16u);
            __builtin_amdgcn_global_load_async_to_lds_b128(g, l, 0, 0);
        }
        __builtin_amdgcn_s_wait_asynccnt(0);
#else
        const uint4* src = (const uint4*)gsrc;
        uint4* dst = (uint4*)&sW[0];
        for (int i = threadIdx.x; i < 81920 / 16; i += 256) dst[i] = src[i];
#endif
        __syncthreads();
    }
    if (tile >= NTILES) return;

    const _Float16* sB1 = sW;
    const _Float16* sB2 = sW + 8192;
    const _Float16* sB3 = sW + 16384;

    int n0   = tile * 16;
    int m    = lane & 15;
    int kb   = (lane < 16) ? 0 : 8;
    int moff = (lane < 16) ? 0 : 8;
    _Float16* feat = sFeat[wave];

    // prefetch the streams needed after GEMM1 (normalfea row, geometry)
    __builtin_prefetch(normalfea + ((size_t)b * NN + n0 + m) * CC, 0, 1);
    __builtin_prefetch(point  + ((size_t)b * NN + n0 + m) * 3, 0, 1);
    __builtin_prefetch(normal + ((size_t)b * NN + n0 + m) * 3, 0, 1);

    // ---------- GEMM1: pf = relu(pointfea[16x128] @ fc1_w[128x64] + b1) ----------
    {
        const float* rowA = pointfea + ((size_t)b * NN + n0 + m) * CC;
        v8f acc[4] = {};
        #pragma unroll
        for (int ks = 0; ks < 4; ++ks) {
            const float* p0 = rowA + ks * 32 + kb;
            v16h a;
            #pragma unroll
            for (int h = 0; h < 8; ++h) { a[h] = (_Float16)p0[h]; a[8 + h] = (_Float16)p0[16 + h]; }
            const _Float16* fb = sB1 + (size_t)(ks * 4) * 512 + lane * 16;
            #pragma unroll
            for (int t = 0; t < 4; ++t) {
                v16h bt = *(const v16h*)(fb + (size_t)t * 512);
                acc[t] = __builtin_amdgcn_wmma_f32_16x16x32_f16(false, a, false, bt,
                                                                (short)0, acc[t], false, false);
            }
        }
        int ch = lane & 15;
        #pragma unroll
        for (int t = 0; t < 4; ++t) {
            float bias = b12[t * 16 + ch];
            #pragma unroll
            for (int r = 0; r < 8; ++r)
                feat[(moff + r) * 192 + (t * 16 + ch)] = (_Float16)fmaxf(acc[t][r] + bias, 0.f);
        }
    }

    // ---------- GEMM2: nf = relu(normalfea @ fc2_w + b2) ----------
    {
        const float* rowA = normalfea + ((size_t)b * NN + n0 + m) * CC;
        v8f acc[4] = {};
        #pragma unroll
        for (int ks = 0; ks < 4; ++ks) {
            const float* p0 = rowA + ks * 32 + kb;
            v16h a;
            #pragma unroll
            for (int h = 0; h < 8; ++h) { a[h] = (_Float16)p0[h]; a[8 + h] = (_Float16)p0[16 + h]; }
            const _Float16* fb = sB2 + (size_t)(ks * 4) * 512 + lane * 16;
            #pragma unroll
            for (int t = 0; t < 4; ++t) {
                v16h bt = *(const v16h*)(fb + (size_t)t * 512);
                acc[t] = __builtin_amdgcn_wmma_f32_16x16x32_f16(false, a, false, bt,
                                                                (short)0, acc[t], false, false);
            }
        }
        int ch = lane & 15;
        #pragma unroll
        for (int t = 0; t < 4; ++t) {
            float bias = b12[64 + t * 16 + ch];
            #pragma unroll
            for (int r = 0; r < 8; ++r)
                feat[(moff + r) * 192 + 64 + (t * 16 + ch)] = (_Float16)fmaxf(acc[t][r] + bias, 0.f);
        }
    }

    // ---------- angle / dist features (fc3, fc4: 1 -> 32 each) ----------
    {
        const float* P = point  + ((size_t)b * NN + n0 + m) * 3;
        const float* Q = normal + ((size_t)b * NN + n0 + m) * 3;
        float px = P[0], py = P[1], pz = P[2];
        float qx = Q[0], qy = Q[1], qz = Q[2];
        float dist = __expf(-(px * px + py * py + pz * pz));
        float cx = py * qz - pz * qy, cy = pz * qx - px * qz, cz = px * qy - py * qx;
        float cn = sqrtf(cx * cx + cy * cy + cz * cz);
        float dt = px * qx + py * qy + pz * qz;
        float ang = atan2f(cn, dt);
        int chb = (lane < 16) ? 0 : 16;
        #pragma unroll
        for (int j = 0; j < 16; ++j) {
            int ch = chb + j;
            feat[m * 192 + 128 + ch] = (_Float16)fmaxf(ang  * fc34[ch]      + fc34[32 + ch], 0.f);
            feat[m * 192 + 160 + ch] = (_Float16)fmaxf(dist * fc34[64 + ch] + fc34[96 + ch], 0.f);
        }
    }

    // ---------- GEMM3: out[16x128] = feature[16x192] @ conv_w^T[192x128] ----------
    v8f accC[8] = {};
    #pragma unroll
    for (int ks = 0; ks < 6; ++ks) {
        const _Float16* fr = feat + m * 192 + ks * 32 + kb;
        v8h lo = *(const v8h*)fr;
        v8h hi = *(const v8h*)(fr + 16);
        v16h a = __builtin_shufflevector(lo, hi, 0, 1, 2, 3, 4, 5, 6, 7,
                                                 8, 9, 10, 11, 12, 13, 14, 15);
        const _Float16* fb = sB3 + (size_t)(ks * 8) * 512 + lane * 16;
        #pragma unroll
        for (int t = 0; t < 8; ++t) {
            v16h bt = *(const v16h*)(fb + (size_t)t * 512);
            accC[t] = __builtin_amdgcn_wmma_f32_16x16x32_f16(false, a, false, bt,
                                                             (short)0, accC[t], false, false);
        }
    }

    // ---------- BN + LeakyReLU + channel-max + sigmoid ----------
    float pm[8];
    #pragma unroll
    for (int r = 0; r < 8; ++r) pm[r] = -3.4e38f;
    int ch0 = lane & 15;
    #pragma unroll
    for (int t = 0; t < 8; ++t) {
        int ch = t * 16 + ch0;
        float sc = bnf[ch], sh = bnf[128 + ch];
        #pragma unroll
        for (int r = 0; r < 8; ++r) {
            float y = accC[t][r] * sc + sh;
            y = (y >= 0.f) ? y : 0.2f * y;
            pm[r] = fmaxf(pm[r], y);
        }
    }
    // butterfly max over the 16-lane half (wave32: xor bits 0..3 stay in half)
    #pragma unroll
    for (int off = 8; off >= 1; off >>= 1) {
        #pragma unroll
        for (int r = 0; r < 8; ++r)
            pm[r] = fmaxf(pm[r], __shfl_xor(pm[r], off, 32));
    }
    if ((lane & 15) == 0) {   // lanes 0 and 16 emit points 0-7 / 8-15
        #pragma unroll
        for (int r = 0; r < 8; ++r) {
            int n = n0 + moff + r;
            float w = 1.f / (1.f + __expf(-pm[r]));
            out_weight[(size_t)b * NN + n] = w;
            keys[(size_t)b * NPAD + n] =
                ((u64)__float_as_uint(w) << 32) | (u32)(0xFFFFFFFFu - (u32)n);
        }
    }
}

// ===================== bitonic sort (descending) =====================
__global__ void bitonic_kernel(u64* __restrict__ keys, int j, int k)
{
    int i = blockIdx.x * blockDim.x + threadIdx.x;
    u64* kb = keys + (size_t)blockIdx.y * NPAD;
    int ixj = i ^ j;
    if (ixj > i) {
        u64 a = kb[i], c = kb[ixj];
        bool desc = ((i & k) == 0);
        bool sw = desc ? (a < c) : (a > c);
        if (sw) { kb[i] = c; kb[ixj] = a; }
    }
}

// ===================== top-K gather =====================
__global__ void gather_kernel(const u64* __restrict__ keys,
                              const float* __restrict__ pointfea,
                              const float* __restrict__ normalfea,
                              const float* __restrict__ point,
                              const float* __restrict__ normal,
                              float* __restrict__ out)
{
    int gid = blockIdx.x * blockDim.x + threadIdx.x;
    int gwave = gid >> 5, lane = gid & 31;
    if (gwave >= BB * KK) return;
    int b = gwave / KK, k = gwave - b * KK;
    u64 kv = keys[(size_t)b * NPAD + k];
    u32 idx = 0xFFFFFFFFu - (u32)(kv & 0xFFFFFFFFu);

    const float4* srcP = (const float4*)(pointfea  + ((size_t)b * NN + idx) * CC);
    const float4* srcN = (const float4*)(normalfea + ((size_t)b * NN + idx) * CC);
    float4* dP = (float4*)(out + OUT_KPF + ((size_t)b * KK + k) * CC);
    float4* dN = (float4*)(out + OUT_KNF + ((size_t)b * KK + k) * CC);
    dP[lane] = srcP[lane];
    dN[lane] = srcN[lane];
    if (lane < 3) {
        out[OUT_KRP + ((size_t)b * KK + k) * 3 + lane] = point [((size_t)b * NN + idx) * 3 + lane];
        out[OUT_KRN + ((size_t)b * KK + k) * 3 + lane] = normal[((size_t)b * NN + idx) * 3 + lane];
    }
    if (lane == 3) {
        ((int*)out)[OUT_IDX + (size_t)b * KK + k] = (int)idx;
    }
}

// ===================== host launch =====================
extern "C" void kernel_launch(void* const* d_in, const int* in_sizes, int n_in,
                              void* d_out, int out_size, void* d_ws, size_t ws_size,
                              hipStream_t stream)
{
    const float* pointfea  = (const float*)d_in[0];
    const float* normalfea = (const float*)d_in[1];
    /* d_in[2] "index" unused by reference */
    const float* point     = (const float*)d_in[3];
    const float* normal    = (const float*)d_in[4];
    const float* fc1_w = (const float*)d_in[5],  *fc1_b = (const float*)d_in[6];
    const float* fc2_w = (const float*)d_in[7],  *fc2_b = (const float*)d_in[8];
    const float* fc3_w = (const float*)d_in[9],  *fc3_b = (const float*)d_in[10];
    const float* fc4_w = (const float*)d_in[11], *fc4_b = (const float*)d_in[12];
    const float* conv_w = (const float*)d_in[13], *conv_b = (const float*)d_in[14];
    const float* bn_g  = (const float*)d_in[15], *bn_be = (const float*)d_in[16];
    const float* bn_mu = (const float*)d_in[17], *bn_var = (const float*)d_in[18];

    char* ws   = (char*)d_ws;
    float* out = (float*)d_out;
    u64* keys  = (u64*)(ws + WS_KEYS);

    prep_kernel<<<1, 256, 0, stream>>>(fc1_w, fc1_b, fc2_w, fc2_b, fc3_w, fc3_b,
                                       fc4_w, fc4_b, conv_w, conv_b,
                                       bn_g, bn_be, bn_mu, bn_var, ws);

    {
        int per = NPAD - NN;
        int tot = BB * per;
        padinit_kernel<<<(tot + 255) / 256, 256, 0, stream>>>(keys);
    }

    score_kernel<<<dim3((NTILES + 7) / 8, BB), 256, 0, stream>>>(
        pointfea, normalfea, point, normal, ws, out + OUT_W, keys);

    for (int k = 2; k <= NPAD; k <<= 1)
        for (int j = k >> 1; j > 0; j >>= 1)
            bitonic_kernel<<<dim3(NPAD / 256, BB), 256, 0, stream>>>(keys, j, k);

    {
        long long threads = (long long)BB * KK * 32;
        gather_kernel<<<(int)((threads + 255) / 256), 256, 0, stream>>>(
            keys, pointfea, normalfea, point, normal, out);
    }
}
